// CausalAttention_88682484728577
// MI455X (gfx1250) — matrix operand order repaired and checked by reading, not compiled, not run on hardware
//
#include <hip/hip_runtime.h>

typedef __bf16 bf16;
typedef __attribute__((ext_vector_type(16))) __bf16 v16bf;
typedef __attribute__((ext_vector_type(8)))  __bf16 v8bf;
typedef __attribute__((ext_vector_type(8)))  float  v8f;
typedef __attribute__((ext_vector_type(4)))  int    v4i;
typedef __attribute__((address_space(3)))    v4i    as3_v4i;

#define WMMA_BF16(a,b,c) \
  __builtin_amdgcn_wmma_f32_16x16x32_bf16(false,(a),false,(b),(short)0,(c),false,false)

#if __has_builtin(__builtin_amdgcn_global_load_async_to_lds_b128) && \
    __has_builtin(__builtin_amdgcn_s_wait_asynccnt)
#define HAVE_ASYNC 1
#endif

// copy 16 bytes global -> LDS (async when available)
static __device__ __forceinline__ void cp16(const bf16* g, bf16* l) {
#ifdef HAVE_ASYNC
  __builtin_amdgcn_global_load_async_to_lds_b128((v4i*)g, (as3_v4i*)l, 0, 0);
#else
  *(v8bf*)l = *(const v8bf*)g;
#endif
}
#ifdef HAVE_ASYNC
#define ASYNC_WAIT(n) __builtin_amdgcn_s_wait_asynccnt(n)
#else
#define ASYNC_WAIT(n)
#endif

static __device__ __forceinline__ v16bf cat16(v8bf lo, v8bf hi) {
  v16bf r;
#pragma unroll
  for (int i = 0; i < 8; i++) { r[i] = lo[i]; r[i + 8] = hi[i]; }
  return r;
}

#define NEG_INF (-__builtin_inff())

// ---------------------------------------------------------------------------
// Stage 0: f32 -> bf16 convert / transpose-convert
// ---------------------------------------------------------------------------
__global__ void cvt_bf16_kernel(const float* __restrict__ in, bf16* __restrict__ out, int n) {
  int i = blockIdx.x * 256 + threadIdx.x;
  if (i < n) out[i] = (bf16)in[i];
}

// out[c][r] = in[r][c];  out shape [cols_in][rows_in]
__global__ void tcvt_bf16_kernel(const float* __restrict__ in, bf16* __restrict__ out,
                                 int rows_in, int cols_in) {
  int i = blockIdx.x * 256 + threadIdx.x;
  if (i < rows_in * cols_in) {
    int c = i / rows_in, r = i % rows_in;
    out[i] = (bf16)in[r * cols_in + c];
  }
}

// ---------------------------------------------------------------------------
// Stage 1: QKV GEMM  [8192 x 1024] x [1024 x 3072] + bias
//   128x128 block tile, 8 waves (each 32x64), double-buffered async LDS.
//   Epilogue scatters to Q[B,H,S,64], K[B,H,S,64], V^T[B,H,64,S]  (bf16)
// ---------------------------------------------------------------------------
__global__ __launch_bounds__(256) void qkv_gemm_kernel(
    const bf16* __restrict__ X, const bf16* __restrict__ Wt,
    const float* __restrict__ bias,
    bf16* __restrict__ Qb, bf16* __restrict__ Kb, bf16* __restrict__ Vt) {
  __shared__ bf16 As[2][128 * 40];
  __shared__ bf16 Bs[2][128 * 40];
  const int m0 = blockIdx.x * 128;
  const int n0 = blockIdx.y * 128;
  const int tid = threadIdx.x;
  const int w = tid >> 5, lane = tid & 31;
  const int n16 = lane & 15, hi = lane >> 4, kb = hi * 8;
  const int wr = w & 3, wc = w >> 2;          // wave sub-tile: rows wr*32, cols wc*64
  const int ldr = tid >> 1, ldc = (tid & 1) * 16;

  v8f acc[2][4];
#pragma unroll
  for (int mt = 0; mt < 2; mt++)
#pragma unroll
    for (int nt = 0; nt < 4; nt++) { v8f z = {}; acc[mt][nt] = z; }

  // prologue: issue k-step 0 into buffer 0
  cp16(&X[(m0 + ldr) * 1024 + ldc],      &As[0][ldr * 40 + ldc]);
  cp16(&X[(m0 + ldr) * 1024 + ldc + 8],  &As[0][ldr * 40 + ldc + 8]);
  cp16(&Wt[(n0 + ldr) * 1024 + ldc],     &Bs[0][ldr * 40 + ldc]);
  cp16(&Wt[(n0 + ldr) * 1024 + ldc + 8], &Bs[0][ldr * 40 + ldc + 8]);

  for (int ks = 0; ks < 32; ks++) {
    const int buf = ks & 1;
    if (ks + 1 < 32) {
      const int k0 = (ks + 1) * 32;
      cp16(&X[(m0 + ldr) * 1024 + k0 + ldc],      &As[buf ^ 1][ldr * 40 + ldc]);
      cp16(&X[(m0 + ldr) * 1024 + k0 + ldc + 8],  &As[buf ^ 1][ldr * 40 + ldc + 8]);
      cp16(&Wt[(n0 + ldr) * 1024 + k0 + ldc],     &Bs[buf ^ 1][ldr * 40 + ldc]);
      cp16(&Wt[(n0 + ldr) * 1024 + k0 + ldc + 8], &Bs[buf ^ 1][ldr * 40 + ldc + 8]);
      ASYNC_WAIT(4);   // 8 outstanding, in-order: <=4 means current buffer done
    } else {
      ASYNC_WAIT(0);
    }
    __syncthreads();

    v16bf a0 = cat16(*(const v8bf*)&As[buf][(wr * 32 + n16) * 40 + hi * 16],
                     *(const v8bf*)&As[buf][(wr * 32 + n16) * 40 + hi * 16 + 8]);
    v16bf a1 = cat16(*(const v8bf*)&As[buf][(wr * 32 + 16 + n16) * 40 + hi * 16],
                     *(const v8bf*)&As[buf][(wr * 32 + 16 + n16) * 40 + hi * 16 + 8]);
#pragma unroll
    for (int nt = 0; nt < 4; nt++) {
      v16bf b = cat16(*(const v8bf*)&Bs[buf][(wc * 64 + nt * 16 + n16) * 40 + hi * 16],
                      *(const v8bf*)&Bs[buf][(wc * 64 + nt * 16 + n16) * 40 + hi * 16 + 8]);
      acc[0][nt] = WMMA_BF16(a0, b, acc[0][nt]);
      acc[1][nt] = WMMA_BF16(a1, b, acc[1][nt]);
    }
    __syncthreads();
  }

#pragma unroll
  for (int nt = 0; nt < 4; nt++) {
    int c = n0 + wc * 64 + nt * 16 + n16;
    float bv = bias[c];
    int which = c >> 10, dcol = c & 1023;
    int h = dcol >> 6, hd = dcol & 63;
#pragma unroll
    for (int mt = 0; mt < 2; mt++)
#pragma unroll
      for (int r = 0; r < 8; r++) {
        int rg = m0 + wr * 32 + mt * 16 + r + 8 * hi;
        int b_ = rg >> 11, s = rg & 2047, bh = b_ * 16 + h;
        float v = acc[mt][nt][r] + bv;
        if (which == 0)      Qb[(bh * 2048 + s) * 64 + hd] = (bf16)v;
        else if (which == 1) Kb[(bh * 2048 + s) * 64 + hd] = (bf16)v;
        else                 Vt[(bh * 64 + hd) * 2048 + s] = (bf16)v;
      }
  }
}

// ---------------------------------------------------------------------------
// Stage 2: causal flash attention.  One block = 128 q rows of one (b,h).
//   8 waves x 16 q rows; KV tiles of 32, double-buffered async; online softmax.
// ---------------------------------------------------------------------------
__global__ __launch_bounds__(256) void flash_attn_kernel(
    const bf16* __restrict__ Qb, const bf16* __restrict__ Kb,
    const bf16* __restrict__ Vt, bf16* __restrict__ Ob) {
  __shared__ bf16 Ks[2][32 * 72];     // [kv][hd]
  __shared__ bf16 Vs[2][64 * 40];     // [hd][kv]  (V^T tile)
  __shared__ bf16 Ps[8][16 * 40];     // per-wave P scratch [q][kv]

  const int q0 = blockIdx.x * 128;
  const int bh = blockIdx.y;
  const int b_ = bh >> 4, h = bh & 15;
  const int tid = threadIdx.x, w = tid >> 5, lane = tid & 31;
  const int n16 = lane & 15, hi = lane >> 4, kb = hi * 8;
  const float scale = 0.125f;  // 1/sqrt(64)

  const bf16* Qhead = Qb + (size_t)bh * 2048 * 64;
  const bf16* Khead = Kb + (size_t)bh * 2048 * 64;
  const bf16* Vhead = Vt + (size_t)bh * 64 * 2048;

  // per-thread cooperative-load coordinates
  const int krow = tid >> 3, kcol = (tid & 7) * 8;   // K tile 32 x 64
  const int vrow = tid >> 2, vcol = (tid & 3) * 8;   // V^T tile 64 x 32

  // Q A-fragments (2 k-steps over hd=64) pinned in registers
  const bf16* qrow = Qhead + (q0 + w * 16 + n16) * 64;
  v16bf aq0 = cat16(*(const v8bf*)(qrow + hi * 16),      *(const v8bf*)(qrow + hi * 16 + 8));
  v16bf aq1 = cat16(*(const v8bf*)(qrow + 32 + hi * 16), *(const v8bf*)(qrow + 32 + hi * 16 + 8));

  float m_i[8], l_i[8];
  v8f o0 = {}, o1 = {}, o2 = {}, o3 = {};
#pragma unroll
  for (int r = 0; r < 8; r++) { m_i[r] = NEG_INF; l_i[r] = 0.0f; }

  const int njt = (q0 >> 5) + 4;  // kv tiles of 32 covering [0, q0+127]

  // prologue: tile 0 into buffer 0
  cp16(&Khead[krow * 64 + kcol], &Ks[0][krow * 72 + kcol]);
  cp16(&Vhead[vrow * 2048 + vcol], &Vs[0][vrow * 40 + vcol]);

  for (int j = 0; j < njt; j++) {
    const int buf = j & 1;
    if (j + 1 < njt) {
      const int kv1 = (j + 1) * 32;
      cp16(&Khead[(kv1 + krow) * 64 + kcol], &Ks[buf ^ 1][krow * 72 + kcol]);
      cp16(&Vhead[vrow * 2048 + kv1 + vcol], &Vs[buf ^ 1][vrow * 40 + vcol]);
      ASYNC_WAIT(2);
    } else {
      ASYNC_WAIT(0);
    }
    __syncthreads();

    const int kv0 = j * 32;
    const bf16* ks = Ks[buf];
    const bf16* vs = Vs[buf];

    // scores: S[16q x 32kv] = Q . K^T   (contraction over hd=64, 2 wmma steps)
    v8f s0 = {}, s1 = {};
#pragma unroll
    for (int hs = 0; hs < 2; hs++) {
      v16bf aq = hs ? aq1 : aq0;
      v16bf bk0 = cat16(*(const v8bf*)&ks[n16 * 72 + hs * 32 + hi * 16],
                        *(const v8bf*)&ks[n16 * 72 + hs * 32 + hi * 16 + 8]);
      v16bf bk1 = cat16(*(const v8bf*)&ks[(16 + n16) * 72 + hs * 32 + hi * 16],
                        *(const v8bf*)&ks[(16 + n16) * 72 + hs * 32 + hi * 16 + 8]);
      s0 = WMMA_BF16(aq, bk0, s0);
      s1 = WMMA_BF16(aq, bk1, s1);
    }

    // online softmax (a C-matrix row spans the 16 lanes of a half-wave)
    float p0[8], p1[8];
#pragma unroll
    for (int r = 0; r < 8; r++) {
      int q = q0 + w * 16 + r + 8 * hi;
      float sa = s0[r] * scale, sb = s1[r] * scale;
      if (kv0 + n16 > q)      sa = NEG_INF;
      if (kv0 + 16 + n16 > q) sb = NEG_INF;

      float rm = fmaxf(sa, sb);
#pragma unroll
      for (int off = 1; off < 16; off <<= 1) rm = fmaxf(rm, __shfl_xor(rm, off, 32));
      float mn = fmaxf(m_i[r], rm);
      float al = (mn == m_i[r]) ? 1.0f : __expf(m_i[r] - mn);
      float ea = (sa == NEG_INF) ? 0.0f : __expf(sa - mn);
      float eb = (sb == NEG_INF) ? 0.0f : __expf(sb - mn);
      float rs = ea + eb;
#pragma unroll
      for (int off = 1; off < 16; off <<= 1) rs += __shfl_xor(rs, off, 32);
      l_i[r] = l_i[r] * al + rs;
      m_i[r] = mn;
      o0[r] *= al; o1[r] *= al; o2[r] *= al; o3[r] *= al;
      p0[r] = ea; p1[r] = eb;
    }

    // C-layout -> A-layout for P via wave-private LDS patch
#pragma unroll
    for (int r = 0; r < 8; r++) {
      Ps[w][(r + 8 * hi) * 40 + n16]      = (bf16)p0[r];
      Ps[w][(r + 8 * hi) * 40 + 16 + n16] = (bf16)p1[r];
    }
    v16bf ap = cat16(*(const v8bf*)&Ps[w][n16 * 40 + hi * 16],
                     *(const v8bf*)&Ps[w][n16 * 40 + hi * 16 + 8]);

    // O += P . V   (contraction over kv=32, 4 hd tiles)
    {
      v16bf bv0 = cat16(*(const v8bf*)&vs[(0 * 16 + n16) * 40 + hi * 16],
                        *(const v8bf*)&vs[(0 * 16 + n16) * 40 + hi * 16 + 8]);
      v16bf bv1 = cat16(*(const v8bf*)&vs[(1 * 16 + n16) * 40 + hi * 16],
                        *(const v8bf*)&vs[(1 * 16 + n16) * 40 + hi * 16 + 8]);
      v16bf bv2 = cat16(*(const v8bf*)&vs[(2 * 16 + n16) * 40 + hi * 16],
                        *(const v8bf*)&vs[(2 * 16 + n16) * 40 + hi * 16 + 8]);
      v16bf bv3 = cat16(*(const v8bf*)&vs[(3 * 16 + n16) * 40 + hi * 16],
                        *(const v8bf*)&vs[(3 * 16 + n16) * 40 + hi * 16 + 8]);
      o0 = WMMA_BF16(ap, bv0, o0);
      o1 = WMMA_BF16(ap, bv1, o1);
      o2 = WMMA_BF16(ap, bv2, o2);
      o3 = WMMA_BF16(ap, bv3, o3);
    }
    __syncthreads();
  }

  // epilogue: O/l -> Ob in [B,S,D] layout (bf16)
#pragma unroll
  for (int r = 0; r < 8; r++) {
    int q = q0 + w * 16 + r + 8 * hi;
    float inv = 1.0f / l_i[r];
    bf16* dst = Ob + ((size_t)(b_ * 2048 + q)) * 1024 + h * 64;
    dst[0 * 16 + n16] = (bf16)(o0[r] * inv);
    dst[1 * 16 + n16] = (bf16)(o1[r] * inv);
    dst[2 * 16 + n16] = (bf16)(o2[r] * inv);
    dst[3 * 16 + n16] = (bf16)(o3[r] * inv);
  }
}

// ---------------------------------------------------------------------------
// Stage 3: output projection  [8192 x 1024] x [1024 x 1024] + bias -> f32 out
// ---------------------------------------------------------------------------
__global__ __launch_bounds__(256) void out_proj_kernel(
    const bf16* __restrict__ A, const bf16* __restrict__ Wt,
    const float* __restrict__ bias, float* __restrict__ out) {
  __shared__ bf16 As[2][128 * 40];
  __shared__ bf16 Bs[2][128 * 40];
  const int m0 = blockIdx.x * 128;
  const int n0 = blockIdx.y * 128;
  const int tid = threadIdx.x;
  const int w = tid >> 5, lane = tid & 31;
  const int n16 = lane & 15, hi = lane >> 4, kb = hi * 8;
  const int wr = w & 3, wc = w >> 2;
  const int ldr = tid >> 1, ldc = (tid & 1) * 16;

  v8f acc[2][4];
#pragma unroll
  for (int mt = 0; mt < 2; mt++)
#pragma unroll
    for (int nt = 0; nt < 4; nt++) { v8f z = {}; acc[mt][nt] = z; }

  cp16(&A[(m0 + ldr) * 1024 + ldc],      &As[0][ldr * 40 + ldc]);
  cp16(&A[(m0 + ldr) * 1024 + ldc + 8],  &As[0][ldr * 40 + ldc + 8]);
  cp16(&Wt[(n0 + ldr) * 1024 + ldc],     &Bs[0][ldr * 40 + ldc]);
  cp16(&Wt[(n0 + ldr) * 1024 + ldc + 8], &Bs[0][ldr * 40 + ldc + 8]);

  for (int ks = 0; ks < 32; ks++) {
    const int buf = ks & 1;
    if (ks + 1 < 32) {
      const int k0 = (ks + 1) * 32;
      cp16(&A[(m0 + ldr) * 1024 + k0 + ldc],      &As[buf ^ 1][ldr * 40 + ldc]);
      cp16(&A[(m0 + ldr) * 1024 + k0 + ldc + 8],  &As[buf ^ 1][ldr * 40 + ldc + 8]);
      cp16(&Wt[(n0 + ldr) * 1024 + k0 + ldc],     &Bs[buf ^ 1][ldr * 40 + ldc]);
      cp16(&Wt[(n0 + ldr) * 1024 + k0 + ldc + 8], &Bs[buf ^ 1][ldr * 40 + ldc + 8]);
      ASYNC_WAIT(4);
    } else {
      ASYNC_WAIT(0);
    }
    __syncthreads();

    v16bf a0 = cat16(*(const v8bf*)&As[buf][(wr * 32 + n16) * 40 + hi * 16],
                     *(const v8bf*)&As[buf][(wr * 32 + n16) * 40 + hi * 16 + 8]);
    v16bf a1 = cat16(*(const v8bf*)&As[buf][(wr * 32 + 16 + n16) * 40 + hi * 16],
                     *(const v8bf*)&As[buf][(wr * 32 + 16 + n16) * 40 + hi * 16 + 8]);
#pragma unroll
    for (int nt = 0; nt < 4; nt++) {
      v16bf b = cat16(*(const v8bf*)&Bs[buf][(wc * 64 + nt * 16 + n16) * 40 + hi * 16],
                      *(const v8bf*)&Bs[buf][(wc * 64 + nt * 16 + n16) * 40 + hi * 16 + 8]);
      acc[0][nt] = WMMA_BF16(a0, b, acc[0][nt]);
      acc[1][nt] = WMMA_BF16(a1, b, acc[1][nt]);
    }
    __syncthreads();
  }

#pragma unroll
  for (int nt = 0; nt < 4; nt++) {
    int c = n0 + wc * 64 + nt * 16 + n16;
    float bv = bias[c];
#pragma unroll
    for (int mt = 0; mt < 2; mt++)
#pragma unroll
      for (int r = 0; r < 8; r++) {
        int rg = m0 + wr * 32 + mt * 16 + r + 8 * hi;
        out[(size_t)rg * 1024 + c] = acc[mt][nt][r] + bv;
      }
  }
}

// ---------------------------------------------------------------------------
extern "C" void kernel_launch(void* const* d_in, const int* in_sizes, int n_in,
                              void* d_out, int out_size, void* d_ws, size_t ws_size,
                              hipStream_t stream) {
  const float* x  = (const float*)d_in[0];
  const float* Wa = (const float*)d_in[1];
  const float* ba = (const float*)d_in[2];
  const float* Wp = (const float*)d_in[3];
  const float* bp = (const float*)d_in[4];
  float* out = (float*)d_out;

  // workspace layout (bf16 elements)
  bf16* Xb  = (bf16*)d_ws;                 // 8192*1024
  bf16* Wat = Xb  + 8192 * 1024;           // 3072*1024 (W_attn^T)
  bf16* Wpt = Wat + 3072 * 1024;           // 1024*1024 (W_proj^T)
  bf16* Qb  = Wpt + 1024 * 1024;           // [B,H,S,64] = 8388608
  bf16* Kb  = Qb  + 8388608;               // [B,H,S,64]
  bf16* Vt  = Kb  + 8388608;               // [B,H,64,S]
  bf16* Ob  = Vt  + 8388608;               // [B,S,D]

  cvt_bf16_kernel<<<(8388608 + 255) / 256, 256, 0, stream>>>(x, Xb, 8388608);
  tcvt_bf16_kernel<<<(3145728 + 255) / 256, 256, 0, stream>>>(Wa, Wat, 1024, 3072);
  tcvt_bf16_kernel<<<(1048576 + 255) / 256, 256, 0, stream>>>(Wp, Wpt, 1024, 1024);

  qkv_gemm_kernel<<<dim3(64, 24), 256, 0, stream>>>(Xb, Wat, ba, Qb, Kb, Vt);
  flash_attn_kernel<<<dim3(16, 64), 256, 0, stream>>>(Qb, Kb, Vt, Ob);
  out_proj_kernel<<<dim3(64, 8), 256, 0, stream>>>(Ob, Wpt, bp, out);
}
